// MySelfAttention_65317862638419
// MI455X (gfx1250) — compile-verified, matrix operand order checked by reading, not run on hardware
//
#include <hip/hip_runtime.h>
#include <hip/hip_bf16.h>
#include <math.h>

typedef __attribute__((ext_vector_type(16))) _Float16 v16h;
typedef __attribute__((ext_vector_type(8)))  _Float16 v8h;
typedef __attribute__((ext_vector_type(8)))  float    v8f;

#define N_TOK 4096
#define DIMC  256
#define HEADS 8
#define HD    32
#define ATTN_SCALE 0.17677669529663689f  // 32^-0.5

// ---------------------------------------------------------------------------
// Wave reductions (wave32)
// ---------------------------------------------------------------------------
__device__ __forceinline__ float red_sum32(float v) {
#pragma unroll
  for (int off = 1; off < 32; off <<= 1) v += __shfl_xor(v, off, 32);
  return v;
}
__device__ __forceinline__ float red_sum16(float v) {
#pragma unroll
  for (int off = 1; off < 16; off <<= 1) v += __shfl_xor(v, off, 32);
  return v;
}
__device__ __forceinline__ float red_max16(float v) {
#pragma unroll
  for (int off = 1; off < 16; off <<= 1) v = fmaxf(v, __shfl_xor(v, off, 32));
  return v;
}

// ---------------------------------------------------------------------------
// WMMA fragment loader.
// Loads a 16x32 f16 tile stored row-major (row stride = `stride` elements)
// into the 16-bit A/B fragment layout for v_wmma_f32_16x16x32_f16:
//   lanes 0-15 : row = lane,    halfs 0-7 -> K 0-7,  halfs 8-15 -> K 16-23
//   lanes 16-31: row = lane-16, halfs 0-7 -> K 8-15, halfs 8-15 -> K 24-31
// Two contiguous 16-byte v8h loads per lane.
// ---------------------------------------------------------------------------
__device__ __forceinline__ v16h load_frag(const _Float16* base, int stride) {
  int lane = threadIdx.x & 31;
  int m    = lane & 15;
  int kb   = (lane < 16) ? 0 : 8;
  const _Float16* p = base + m * stride + kb;
  v8h lo = *(const v8h*)(p);
  v8h hi = *(const v8h*)(p + 16);
  v16h r;
#pragma unroll
  for (int i = 0; i < 8; ++i) { r[i] = lo[i]; r[i + 8] = hi[i]; }
  return r;
}

// 16x64 output tile per wave: 4 accumulators share one A fragment per k-step.
// C[16, 64] = A[M,K] tile(mT) @ Wt[N,K]^T tiles(nT64*4 .. nT64*4+3)
template <int KDIM>
__device__ __forceinline__ void wmma_gemm_acc4(const _Float16* __restrict__ A,
                                               const _Float16* __restrict__ Wt,
                                               int mT, int nT64, v8f acc[4]) {
#pragma unroll
  for (int kk = 0; kk < KDIM; kk += 32) {
    v16h a = load_frag(A + mT * 16 * KDIM + kk, KDIM);
#pragma unroll
    for (int nn = 0; nn < 4; ++nn) {
      v16h b = load_frag(Wt + (nT64 * 4 + nn) * 16 * KDIM + kk, KDIM);
      acc[nn] = __builtin_amdgcn_wmma_f32_16x16x32_f16(false, a, false, b,
                                                       (short)0, acc[nn], false, false);
    }
  }
}

// ---------------------------------------------------------------------------
// Weight transpose + fp32 -> f16:  w[K,N] -> wt[N,K]
// ---------------------------------------------------------------------------
__global__ void wt_kernel(const float* __restrict__ w, _Float16* __restrict__ wt,
                          int K, int N) {
  int idx = blockIdx.x * 256 + threadIdx.x;
  if (idx < K * N) {
    int kIdx = idx / N, nIdx = idx % N;
    wt[nIdx * K + kIdx] = (_Float16)w[idx];
  }
}

// ---------------------------------------------------------------------------
// LN1 (shared between x and flow) + NCHW -> [n,c] transpose.
// grid (4096, 2), block 32: one wave per token; y=0 -> x, y=1 -> flow.
// ---------------------------------------------------------------------------
__global__ void ln1_kernel(const float* __restrict__ x, const float* __restrict__ flow,
                           const float* __restrict__ g, const float* __restrict__ b,
                           float* __restrict__ xt, _Float16* __restrict__ xn,
                           _Float16* __restrict__ fn) {
  int p = blockIdx.x;
  int lane = threadIdx.x;
  bool isX = (blockIdx.y == 0);
  const float* src = isX ? x : flow;
  float vals[8];
  float s = 0.f;
#pragma unroll
  for (int i = 0; i < 8; ++i) {
    int c = lane + 32 * i;
    vals[i] = src[c * N_TOK + p];
    s += vals[i];
  }
  float mean = red_sum32(s) * (1.f / DIMC);
  float ss = 0.f;
#pragma unroll
  for (int i = 0; i < 8; ++i) { float d = vals[i] - mean; ss += d * d; }
  float rstd = rsqrtf(red_sum32(ss) * (1.f / DIMC) + 1e-5f);
  _Float16* dst = isX ? xn : fn;
#pragma unroll
  for (int i = 0; i < 8; ++i) {
    int c = lane + 32 * i;
    dst[p * DIMC + c] = (_Float16)((vals[i] - mean) * rstd * g[c] + b[c]);
    if (isX) xt[p * DIMC + c] = vals[i];
  }
}

// LN2 on contiguous [n,c] f32 input. grid 4096, block 32.
__global__ void ln2_kernel(const float* __restrict__ x2,
                           const float* __restrict__ g, const float* __restrict__ b,
                           _Float16* __restrict__ xn2) {
  int p = blockIdx.x;
  int lane = threadIdx.x;
  float vals[8];
  float s = 0.f;
#pragma unroll
  for (int i = 0; i < 8; ++i) {
    vals[i] = x2[p * DIMC + i * 32 + lane];
    s += vals[i];
  }
  float mean = red_sum32(s) * (1.f / DIMC);
  float ss = 0.f;
#pragma unroll
  for (int i = 0; i < 8; ++i) { float d = vals[i] - mean; ss += d * d; }
  float rstd = rsqrtf(red_sum32(ss) * (1.f / DIMC) + 1e-5f);
#pragma unroll
  for (int i = 0; i < 8; ++i) {
    int c = i * 32 + lane;
    xn2[p * DIMC + c] = (_Float16)((vals[i] - mean) * rstd * g[c] + b[c]);
  }
}

// ---------------------------------------------------------------------------
// QV projection: qv = xn @ w_qv + b_qv, split into Q (scaled) and V^T.
// grid (256, 8), block 32: 16x64 tile per wave.
// ---------------------------------------------------------------------------
__global__ void qv_kernel(const _Float16* __restrict__ xn, const _Float16* __restrict__ wqvT,
                          const float* __restrict__ bqv,
                          _Float16* __restrict__ q, _Float16* __restrict__ vT) {
  int mT = blockIdx.x, nT64 = blockIdx.y;
  v8f acc[4] = {};
  wmma_gemm_acc4<DIMC>(xn, wqvT, mT, nT64, acc);
  int lane = threadIdx.x;
  int col = lane & 15, rb = (lane >> 4) * 8;
#pragma unroll
  for (int nn = 0; nn < 4; ++nn) {
    int j = (nT64 * 4 + nn) * 16 + col;
    int sel = j >> 8, head = (j >> 5) & 7, dd = j & 31;
    float bias = bqv[j];
#pragma unroll
    for (int r = 0; r < 8; ++r) {
      int row = mT * 16 + rb + r;
      float val = acc[nn][r] + bias;
      if (sel == 0) q[(head * N_TOK + row) * HD + dd] = (_Float16)(val * ATTN_SCALE);
      else          vT[(head * HD + dd) * N_TOK + row] = (_Float16)val;
    }
  }
}

// K projection: k = fn @ w_k + b_k -> [H, n, hd]. grid (256, 4), block 32.
__global__ void k_kernel(const _Float16* __restrict__ fn, const _Float16* __restrict__ wkT,
                         const float* __restrict__ bk, _Float16* __restrict__ kbuf) {
  int mT = blockIdx.x, nT64 = blockIdx.y;
  v8f acc[4] = {};
  wmma_gemm_acc4<DIMC>(fn, wkT, mT, nT64, acc);
  int lane = threadIdx.x;
  int col = lane & 15, rb = (lane >> 4) * 8;
#pragma unroll
  for (int nn = 0; nn < 4; ++nn) {
    int j = (nT64 * 4 + nn) * 16 + col;
    int head = j >> 5, dd = j & 31;
    float bias = bk[j];
#pragma unroll
    for (int r = 0; r < 8; ++r) {
      int row = mT * 16 + rb + r;
      kbuf[(head * N_TOK + row) * HD + dd] = (_Float16)(acc[nn][r] + bias);
    }
  }
}

// ---------------------------------------------------------------------------
// Flash attention: one wave per (16-query tile, head). grid (256, 8), block 32.
// Per 32-key step: 2 score WMMAs -> online softmax -> P via LDS -> 2 PV WMMAs.
// ---------------------------------------------------------------------------
__global__ void attn_kernel(const _Float16* __restrict__ q, const _Float16* __restrict__ k,
                            const _Float16* __restrict__ vT, _Float16* __restrict__ ao) {
  __shared__ _Float16 P[16 * 32];
  int mT = blockIdx.x, head = blockIdx.y;
  int lane = threadIdx.x;
  const _Float16* qh = q  + (size_t)head * N_TOK * HD;
  const _Float16* kh = k  + (size_t)head * N_TOK * HD;
  const _Float16* vh = vT + (size_t)head * HD * N_TOK;

  v16h qa = load_frag(qh + mT * 16 * HD, HD);
  v8f o0 = {}, o1 = {};
  float mrun[8], lrun[8];
#pragma unroll
  for (int r = 0; r < 8; ++r) { mrun[r] = -1e30f; lrun[r] = 0.f; }
  int col = lane & 15, rowb = (lane >> 4) * 8;

  for (int j = 0; j < N_TOK; j += 32) {
    // Prefetch next key/value block into cache while we do softmax VALU work.
    if (j + 32 < N_TOK) {
      __builtin_prefetch(kh + (j + 32) * HD + lane * 4, 0, 1);
      __builtin_prefetch(vh + (j + 32) + lane * N_TOK, 0, 1);
    }
    v16h kb0 = load_frag(kh + (j +  0) * HD, HD);
    v16h kb1 = load_frag(kh + (j + 16) * HD, HD);
    v8f z = {};
    v8f s0 = __builtin_amdgcn_wmma_f32_16x16x32_f16(false, qa, false, kb0, (short)0, z, false, false);
    v8f s1 = __builtin_amdgcn_wmma_f32_16x16x32_f16(false, qa, false, kb1, (short)0, z, false, false);

#pragma unroll
    for (int r = 0; r < 8; ++r) {
      float sm   = red_max16(fmaxf(s0[r], s1[r]));
      float mnew = fmaxf(mrun[r], sm);
      float p0   = __expf(s0[r] - mnew);
      float p1   = __expf(s1[r] - mnew);
      float ps   = red_sum16(p0 + p1);
      float corr = __expf(mrun[r] - mnew);
      lrun[r] = lrun[r] * corr + ps;
      mrun[r] = mnew;
      o0[r] *= corr;
      o1[r] *= corr;
      int row = rowb + r;
      P[row * 32 + col]      = (_Float16)p0;
      P[row * 32 + 16 + col] = (_Float16)p1;
    }
    __syncthreads();  // single-wave WG: cheap fence for LDS D->A relayout

    v16h pa  = load_frag(P, 32);
    v16h vb0 = load_frag(vh + j, N_TOK);               // head dims 0-15
    v16h vb1 = load_frag(vh + 16 * N_TOK + j, N_TOK);  // head dims 16-31
    o0 = __builtin_amdgcn_wmma_f32_16x16x32_f16(false, pa, false, vb0, (short)0, o0, false, false);
    o1 = __builtin_amdgcn_wmma_f32_16x16x32_f16(false, pa, false, vb1, (short)0, o1, false, false);
  }

#pragma unroll
  for (int r = 0; r < 8; ++r) {
    float inv = 1.f / lrun[r];
    int row = mT * 16 + rowb + r;
    ao[row * DIMC + head * HD + col]      = (_Float16)(o0[r] * inv);
    ao[row * DIMC + head * HD + 16 + col] = (_Float16)(o1[r] * inv);
  }
}

// ---------------------------------------------------------------------------
// proj + residual: x2 = shortcut + ao @ w_proj + b. grid (256, 4), block 32.
// ---------------------------------------------------------------------------
__global__ void proj_kernel(const _Float16* __restrict__ ao, const _Float16* __restrict__ wT,
                            const float* __restrict__ bias, const float* __restrict__ xt,
                            float* __restrict__ x2) {
  int mT = blockIdx.x, nT64 = blockIdx.y;
  v8f acc[4] = {};
  wmma_gemm_acc4<DIMC>(ao, wT, mT, nT64, acc);
  int lane = threadIdx.x;
  int col = lane & 15, rb = (lane >> 4) * 8;
#pragma unroll
  for (int nn = 0; nn < 4; ++nn) {
    int cg = (nT64 * 4 + nn) * 16 + col;
    float bv = bias[cg];
#pragma unroll
    for (int r = 0; r < 8; ++r) {
      int row = mT * 16 + rb + r;
      x2[row * DIMC + cg] = xt[row * DIMC + cg] + acc[nn][r] + bv;
    }
  }
}

// MLP1 + exact GELU. grid (256, 8), block 32.
__global__ void mlp1_kernel(const _Float16* __restrict__ xn2, const _Float16* __restrict__ wT,
                            const float* __restrict__ bias, _Float16* __restrict__ h) {
  int mT = blockIdx.x, nT64 = blockIdx.y;
  v8f acc[4] = {};
  wmma_gemm_acc4<DIMC>(xn2, wT, mT, nT64, acc);
  int lane = threadIdx.x;
  int col = lane & 15, rb = (lane >> 4) * 8;
#pragma unroll
  for (int nn = 0; nn < 4; ++nn) {
    int cg = (nT64 * 4 + nn) * 16 + col;
    float bv = bias[cg];
#pragma unroll
    for (int r = 0; r < 8; ++r) {
      int row = mT * 16 + rb + r;
      float val = acc[nn][r] + bv;
      float gel = 0.5f * val * (1.f + erff(val * 0.70710678118654752f));
      h[row * (2 * DIMC) + cg] = (_Float16)gel;
    }
  }
}

// MLP2 + residual; epilogue writes final [c,h,w] layout. grid (256, 4), block 32.
__global__ void mlp2_kernel(const _Float16* __restrict__ h, const _Float16* __restrict__ wT,
                            const float* __restrict__ bias, const float* __restrict__ x2,
                            float* __restrict__ out) {
  int mT = blockIdx.x, nT64 = blockIdx.y;
  v8f acc[4] = {};
  wmma_gemm_acc4<2 * DIMC>(h, wT, mT, nT64, acc);
  int lane = threadIdx.x;
  int col = lane & 15, rb = (lane >> 4) * 8;
#pragma unroll
  for (int nn = 0; nn < 4; ++nn) {
    int cg = (nT64 * 4 + nn) * 16 + col;
    float bv = bias[cg];
#pragma unroll
    for (int r = 0; r < 8; ++r) {
      int row = mT * 16 + rb + r;
      out[cg * N_TOK + row] = x2[row * DIMC + cg] + acc[nn][r] + bv;  // [c, h*w]
    }
  }
}

// ---------------------------------------------------------------------------
extern "C" void kernel_launch(void* const* d_in, const int* in_sizes, int n_in,
                              void* d_out, int out_size, void* d_ws, size_t ws_size,
                              hipStream_t stream) {
  const float* x      = (const float*)d_in[0];
  const float* flow   = (const float*)d_in[1];
  const float* ln1_g  = (const float*)d_in[2];
  const float* ln1_b  = (const float*)d_in[3];
  const float* w_qv   = (const float*)d_in[4];
  const float* b_qv   = (const float*)d_in[5];
  const float* w_k    = (const float*)d_in[6];
  const float* b_k    = (const float*)d_in[7];
  const float* w_proj = (const float*)d_in[8];
  const float* b_proj = (const float*)d_in[9];
  const float* ln2_g  = (const float*)d_in[10];
  const float* ln2_b  = (const float*)d_in[11];
  const float* w_mlp1 = (const float*)d_in[12];
  const float* b_mlp1 = (const float*)d_in[13];
  const float* w_mlp2 = (const float*)d_in[14];
  const float* b_mlp2 = (const float*)d_in[15];
  float* out = (float*)d_out;

  // Workspace carve-out (all 256B aligned sizes).
  char* ws = (char*)d_ws;
  size_t off = 0;
  auto take = [&](size_t bytes) { char* p = ws + off; off += (bytes + 255) & ~(size_t)255; return p; };
  float*    xt      = (float*)   take((size_t)N_TOK * DIMC * 4);
  _Float16* xn      = (_Float16*)take((size_t)N_TOK * DIMC * 2);
  _Float16* fn      = (_Float16*)take((size_t)N_TOK * DIMC * 2);
  _Float16* wqvT    = (_Float16*)take((size_t)2 * DIMC * DIMC * 2);
  _Float16* wkT     = (_Float16*)take((size_t)DIMC * DIMC * 2);
  _Float16* wprojT  = (_Float16*)take((size_t)DIMC * DIMC * 2);
  _Float16* wmlp1T  = (_Float16*)take((size_t)2 * DIMC * DIMC * 2);
  _Float16* wmlp2T  = (_Float16*)take((size_t)2 * DIMC * DIMC * 2);
  _Float16* qbuf    = (_Float16*)take((size_t)N_TOK * DIMC * 2);
  _Float16* vTbuf   = (_Float16*)take((size_t)N_TOK * DIMC * 2);
  _Float16* kbuf    = (_Float16*)take((size_t)N_TOK * DIMC * 2);
  _Float16* aobuf   = (_Float16*)take((size_t)N_TOK * DIMC * 2);
  float*    x2      = (float*)   take((size_t)N_TOK * DIMC * 4);
  _Float16* xn2     = (_Float16*)take((size_t)N_TOK * DIMC * 2);
  _Float16* hbuf    = (_Float16*)take((size_t)N_TOK * 2 * DIMC * 2);
  (void)ws_size; (void)in_sizes; (void)n_in; (void)out_size;

  // 1. Weight transposes (fp32 [K,N] -> f16 [N,K]).
  wt_kernel<<<dim3((DIMC * 2 * DIMC + 255) / 256), 256, 0, stream>>>(w_qv,   wqvT,   DIMC, 2 * DIMC);
  wt_kernel<<<dim3((DIMC * DIMC     + 255) / 256), 256, 0, stream>>>(w_k,    wkT,    DIMC, DIMC);
  wt_kernel<<<dim3((DIMC * DIMC     + 255) / 256), 256, 0, stream>>>(w_proj, wprojT, DIMC, DIMC);
  wt_kernel<<<dim3((DIMC * 2 * DIMC + 255) / 256), 256, 0, stream>>>(w_mlp1, wmlp1T, DIMC, 2 * DIMC);
  wt_kernel<<<dim3((2 * DIMC * DIMC + 255) / 256), 256, 0, stream>>>(w_mlp2, wmlp2T, 2 * DIMC, DIMC);

  // 2. LN1 for x and flow (+ transpose to [n,c], keep f32 shortcut).
  ln1_kernel<<<dim3(N_TOK, 2), 32, 0, stream>>>(x, flow, ln1_g, ln1_b, xt, xn, fn);

  // 3. Projections (16x64 tile per wave).
  qv_kernel<<<dim3(N_TOK / 16, 2 * DIMC / 64), 32, 0, stream>>>(xn, wqvT, b_qv, qbuf, vTbuf);
  k_kernel <<<dim3(N_TOK / 16, DIMC / 64),     32, 0, stream>>>(fn, wkT,  b_k,  kbuf);

  // 4. Flash attention.
  attn_kernel<<<dim3(N_TOK / 16, HEADS), 32, 0, stream>>>(qbuf, kbuf, vTbuf, aobuf);

  // 5. Output proj + residual.
  proj_kernel<<<dim3(N_TOK / 16, DIMC / 64), 32, 0, stream>>>(aobuf, wprojT, b_proj, xt, x2);

  // 6. LN2.
  ln2_kernel<<<dim3(N_TOK), 32, 0, stream>>>(x2, ln2_g, ln2_b, xn2);

  // 7. MLP.
  mlp1_kernel<<<dim3(N_TOK / 16, 2 * DIMC / 64), 32, 0, stream>>>(xn2, wmlp1T, b_mlp1, hbuf);
  mlp2_kernel<<<dim3(N_TOK / 16, DIMC / 64),     32, 0, stream>>>(hbuf, wmlp2T, b_mlp2, x2, out);
}
